// VectorQuantizer_80599356277310
// MI455X (gfx1250) — compile-verified, matrix operand order checked by reading, not run on hardware
//
#include <hip/hip_runtime.h>

typedef _Float16 f16;
typedef _Float16 v16h __attribute__((ext_vector_type(16)));
typedef float    v8f  __attribute__((ext_vector_type(8)));
typedef unsigned int u32x4 __attribute__((ext_vector_type(4)));
typedef int          i32x8 __attribute__((ext_vector_type(8)));
typedef int          i32x4 __attribute__((ext_vector_type(4)));

#define NUM_EMB 8192
#define DIM     256
#define HW      1024      // 32*32
#define NVEC    16384     // 16 * HW
#define TM      128       // rows per block
#define TK      64        // codes per tile
#define NTILE   (NUM_EMB / TK)          // 128 tiles
#define TILE_U32 (4 * 8 * 32 * 8)       // 8192 u32 = 32 KB per tile image
#define BETA    0.25f

static __device__ __forceinline__ unsigned int pack2(f16 lo, f16 hi) {
  union { f16 h; unsigned short u; } a, b;
  a.h = lo; b.h = hi;
  return ((unsigned int)b.u << 16) | (unsigned int)a.u;
}

// ---------------- Kernel 1a: code norms ||e_k||^2, zero loss accumulator ----
__global__ void __launch_bounds__(256) vq_prep(const float* __restrict__ emb,
                                               float* __restrict__ enorm,
                                               float* loss_acc) {
  const int wv   = threadIdx.x >> 5;
  const int lane = threadIdx.x & 31;
  const int k    = blockIdx.x * 8 + wv;          // 1024 blocks * 8 waves
  const float* row = emb + (long)k * DIM;
  float s = 0.f;
  #pragma unroll
  for (int i = 0; i < DIM; i += 32) {
    float v = row[i + lane];
    s += v * v;
  }
  #pragma unroll
  for (int m = 16; m > 0; m >>= 1) s += __shfl_xor(s, m, 32);
  if (lane == 0) enorm[k] = s;
  if (blockIdx.x == 0 && threadIdx.x == 0) *loss_acc = 0.f;
}

// ---------------- Kernel 1b: pre-tile emb f32 -> f16 B-fragment images ------
// embB[tile] is the exact 32 KB LDS image: [kt(4)][kk(8)][lane(32)][v(8)] u32
__global__ void __launch_bounds__(256) vq_retile(const float* __restrict__ emb,
                                                 unsigned int* __restrict__ embB) {
  const int tid  = threadIdx.x;
  const int tile = blockIdx.x;                   // 128 tiles of 64 codes
  unsigned int* dst = embB + (long)tile * TILE_U32;
  #pragma unroll 4
  for (int i = 0; i < 16; ++i) {
    const int f   = i * 256 + tid;               // float4 index in [0,4096)
    const int row = f >> 4;                      // row offset 0..255
    const int c0  = (f & 15) * 4;                // dim 0..60
    const float4 s4 = *(const float4*)(emb + ((long)tile * TK + row) * DIM + c0);
    const unsigned int w0 = pack2((f16)s4.x, (f16)s4.y);
    const unsigned int w1 = pack2((f16)s4.z, (f16)s4.w);
    const int kk    = c0 >> 5;
    const int rem   = c0 & 31;
    const int bhi   = (rem >> 3) & 1;
    const int v0    = ((rem & 16) ? 4 : 0) + ((rem & 4) ? 2 : 0);
    const int blane = bhi * 16 + (row & 15);
    const int kt    = row >> 4;
    unsigned int* p = dst + ((((kt * 8 + kk) * 32) + blane) * 8 + v0);
    p[0] = w0; p[1] = w1;
  }
}

// Issue one TDM load: contiguous 32 KB (4096 x 8B) global -> LDS byte offset.
static __device__ __forceinline__ void tdm_tile_load(const unsigned int* gsrc,
                                                     unsigned int lds_byte_off) {
  const unsigned long long ga = (unsigned long long)gsrc;
  u32x4 g0;
  g0[0] = 1u;                                    // count=1, no gather
  g0[1] = lds_byte_off;                          // lds_addr
  g0[2] = (unsigned int)ga;                      // global_addr[31:0]
  g0[3] = (unsigned int)(ga >> 32) | 0x80000000u; // global_addr[56:32] | type=2
  i32x8 g1;
  g1[0] = (int)(3u << 16);                       // data_size=8B, mask=0
  g1[1] = (int)(4096u << 16);                    // tensor_dim0 = 4096 (lo16)
  g1[2] = (int)(1u << 16);                       // tensor_dim0 hi=0, tensor_dim1=1
  g1[3] = (int)(4096u << 16);                    // tile_dim0 = 4096
  g1[4] = 1;                                     // tile_dim1 = 1
  g1[5] = 4096;                                  // tensor_dim0_stride lo32
  g1[6] = (int)(4096u << 16);                    // dim0_stride hi=0, dim1_stride lo16
  g1[7] = 0;
  i32x4 g2; g2[0] = 0; g2[1] = 0; g2[2] = 0; g2[3] = 0;
  i32x4 g3; g3[0] = 0; g3[1] = 0; g3[2] = 0; g3[3] = 0;
  i32x8 g4; g4[0] = 0; g4[1] = 0; g4[2] = 0; g4[3] = 0;
              g4[4] = 0; g4[5] = 0; g4[6] = 0; g4[7] = 0;
  __builtin_amdgcn_tensor_load_to_lds(g0, g1, g2, g3, g4, 0);
}

// ---------------- Kernel 2: WMMA GEMM + argmin, TDM double-buffered ---------
// Block: 128 rows x all 8192 codes. Waves split 2-D:
//   wm = wave&3  -> 32-row quad (2 register-resident M-tiles, RM=2)
//   wq = wave>>2 -> half of the 64-code tile (RK=2)
__global__ void __launch_bounds__(256) vq_main(const float* __restrict__ z,
                                               const float* __restrict__ emb,
                                               const unsigned int* __restrict__ embB,
                                               const float* __restrict__ enorm,
                                               float* __restrict__ out_zq,
                                               float* __restrict__ out_idx,
                                               float* loss_acc) {
  __shared__ unsigned int lds[2 * TILE_U32];     // 64 KB, two tile buffers

  const int tid  = threadIdx.x;
  const int wv   = tid >> 5;
  const int lane = tid & 31;
  const int hi   = lane >> 4;
  const int m    = lane & 15;
  const int wm   = wv & 3;            // row quad 0..3
  const int wq   = wv >> 2;           // code half 0..1
  const int kt0  = wq * 2;

  const int n0  = blockIdx.x * TM;
  const int nw  = n0 + wm * 32;
  const int b   = n0 >> 10;
  const int hw0 = nw & 1023;
  const long zbase = (long)b * (DIM * HW) + hw0 + m;

  // ---- A fragments: 2 M-tiles x 256 dims, f32->f16, register resident ----
  v16h a[2][8];
  #pragma unroll
  for (int t = 0; t < 2; ++t) {
    #pragma unroll
    for (int kk = 0; kk < 8; ++kk) {
      #pragma unroll
      for (int e = 0; e < 16; ++e) {
        const int khalf = hi * 8 + e + ((e >= 8) ? 8 : 0);  // 16-bit A layout
        const int c = kk * 32 + khalf;
        a[t][kk][e] = (f16)z[zbase + t * 16 + (long)c * HW];
      }
    }
  }

  float best_d[2][8];
  int   best_i[2][8];
  #pragma unroll
  for (int t = 0; t < 2; ++t)
    #pragma unroll
    for (int v = 0; v < 8; ++v) { best_d[t][v] = 3.0e38f; best_i[t][v] = 0; }

  // prologue: DMA tile 0 into buffer 0
  if (wv == 0) {
    tdm_tile_load(embB, 0u);
    __builtin_amdgcn_s_wait_tensorcnt((short)0);
  }
  __syncthreads();

  for (int it = 0; it < NTILE; ++it) {
    // kick DMA for next tile into the other buffer (overlaps compute via TENSORcnt)
    if (wv == 0 && it + 1 < NTILE) {
      tdm_tile_load(embB + (long)(it + 1) * TILE_U32,
                    (unsigned int)(((it + 1) & 1) * (TILE_U32 * 4)));
    }
    const unsigned int* buf = lds + (it & 1) * TILE_U32;

    // ---- 32 rows x 32 codes via 32 WMMAs, B frags double-buffered ----
    v8f acc[2][2];
    #pragma unroll
    for (int t = 0; t < 2; ++t)
      #pragma unroll
      for (int j = 0; j < 2; ++j)
        #pragma unroll
        for (int e = 0; e < 8; ++e) acc[t][j][e] = 0.f;

    v16h b0 = *(const v16h*)&buf[((kt0 * 8 + 0) * 32 + lane) * 8];
    v16h b1 = *(const v16h*)&buf[(((kt0 + 1) * 8 + 0) * 32 + lane) * 8];
    #pragma unroll
    for (int kk = 0; kk < 8; ++kk) {
      v16h p0, p1;
      if (kk < 7) {
        p0 = *(const v16h*)&buf[((kt0 * 8 + kk + 1) * 32 + lane) * 8];
        p1 = *(const v16h*)&buf[(((kt0 + 1) * 8 + kk + 1) * 32 + lane) * 8];
      }
      acc[0][0] = __builtin_amdgcn_wmma_f32_16x16x32_f16(
          false, a[0][kk], false, b0, (short)0, acc[0][0], false, false);
      acc[1][0] = __builtin_amdgcn_wmma_f32_16x16x32_f16(
          false, a[1][kk], false, b0, (short)0, acc[1][0], false, false);
      acc[0][1] = __builtin_amdgcn_wmma_f32_16x16x32_f16(
          false, a[0][kk], false, b1, (short)0, acc[0][1], false, false);
      acc[1][1] = __builtin_amdgcn_wmma_f32_16x16x32_f16(
          false, a[1][kk], false, b1, (short)0, acc[1][1], false, false);
      if (kk < 7) { b0 = p0; b1 = p1; }
    }

    // ---- epilogue: d = ||e||^2 - 2 z.e ; running argmin ----
    const int kb = it * TK;
    #pragma unroll
    for (int j = 0; j < 2; ++j) {
      const int k  = kb + wq * 32 + j * 16 + m;
      const float nk = enorm[k];
      #pragma unroll
      for (int t = 0; t < 2; ++t) {
        #pragma unroll
        for (int v = 0; v < 8; ++v) {
          const float d = nk - 2.0f * acc[t][j][v];
          if (d < best_d[t][v]) { best_d[t][v] = d; best_i[t][v] = k; }
        }
      }
    }

    if (wv == 0) __builtin_amdgcn_s_wait_tensorcnt((short)0);
    __syncthreads();
  }

  // ---- reduce argmin across the 16 lanes sharing each row (ties -> low idx)
  #pragma unroll
  for (int mask = 1; mask < 16; mask <<= 1) {
    #pragma unroll
    for (int t = 0; t < 2; ++t) {
      #pragma unroll
      for (int v = 0; v < 8; ++v) {
        const float od = __shfl_xor(best_d[t][v], mask, 32);
        const int   oi = __shfl_xor(best_i[t][v], mask, 32);
        if (od < best_d[t][v] || (od == best_d[t][v] && oi < best_i[t][v])) {
          best_d[t][v] = od; best_i[t][v] = oi;
        }
      }
    }
  }

  // tail scratch overlaid on the (now dead) tile buffers
  int*   ldsIdx = (int*)lds;               // [128]
  float* ldsD   = (float*)(lds + 128);     // [2][128]
  int*   ldsI   = (int*)(lds + 384);       // [2][128]
  float* ldsRed = (float*)(lds + 640);     // [256]

  if (m == 0) {
    #pragma unroll
    for (int t = 0; t < 2; ++t)
      #pragma unroll
      for (int v = 0; v < 8; ++v) {
        const int rl = wm * 32 + t * 16 + v + 8 * hi;   // 0..127
        ldsD[wq * TM + rl] = best_d[t][v];
        ldsI[wq * TM + rl] = best_i[t][v];
      }
  }
  __syncthreads();

  if (tid < TM) {
    const float d0 = ldsD[tid], d1 = ldsD[TM + tid];
    const int   i0 = ldsI[tid], i1 = ldsI[TM + tid];
    const int   w  = (d1 < d0 || (d1 == d0 && i1 < i0)) ? i1 : i0;
    ldsIdx[tid] = w;
    out_idx[n0 + tid] = (float)w;
  }
  __syncthreads();

  // ---- gather z_q (f32 emb), write [B,C,H,W] coalesced along HW, loss sum
  const long obase = (long)b * (DIM * HW) + (n0 & 1023);
  float s = 0.f;
  for (int itr = 0; itr < 128; ++itr) {
    const int flat = itr * 256 + tid;      // 256 dims x 128 rows
    const int c = flat >> 7;
    const int r = flat & 127;
    const float e = emb[(long)ldsIdx[r] * DIM + c];
    const long off = obase + (long)c * HW + r;
    out_zq[off] = e;
    const float dv = z[off] - e;
    s += dv * dv;
  }
  ldsRed[tid] = s;
  __syncthreads();
  for (int st = 128; st > 0; st >>= 1) {
    if (tid < st) ldsRed[tid] += ldsRed[tid + st];
    __syncthreads();
  }
  if (tid == 0) atomicAdd(loss_acc, ldsRed[0]);
}

// ---------------- Kernel 3: finalize vq_loss = (1+beta) * mean --------------
__global__ void vq_finish(const float* loss_acc, float* out_loss) {
  *out_loss = (1.0f + BETA) * (*loss_acc) / (float)((long)NVEC * DIM);
}

extern "C" void kernel_launch(void* const* d_in, const int* in_sizes, int n_in,
                              void* d_out, int out_size, void* d_ws, size_t ws_size,
                              hipStream_t stream) {
  const float* z   = (const float*)d_in[0];   // [16,256,32,32] f32
  const float* emb = (const float*)d_in[1];   // [8192,256] f32
  float* out = (float*)d_out;

  float*        enorm    = (float*)d_ws;                 // 8192 f32
  float*        loss_acc = enorm + NUM_EMB;              // 1 f32
  unsigned int* embB     = (unsigned int*)(enorm + 8448); // 128 * 32 KB = 4 MB

  float* out_zq   = out;                          // 4,194,304 f32
  float* out_loss = out + (long)NVEC * DIM;       // 1 f32
  float* out_idx  = out_loss + 1;                 // 16,384 (idx as f32)

  vq_prep  <<<NUM_EMB / 8, 256, 0, stream>>>(emb, enorm, loss_acc);
  vq_retile<<<NTILE,       256, 0, stream>>>(emb, embB);
  vq_main  <<<NVEC / TM,   256, 0, stream>>>(z, emb, embB, enorm,
                                             out_zq, out_idx, loss_acc);
  vq_finish<<<1, 1, 0, stream>>>(loss_acc, out_loss);
}
